// Seq2Seq_28638841930023
// MI455X (gfx1250) — matrix operand order repaired and checked by reading, not compiled, not run on hardware
//
#include <hip/hip_runtime.h>
#include <math.h>

// ---------------- CDNA5 WMMA types ----------------
typedef __attribute__((ext_vector_type(16))) __bf16 v16bf;
typedef __attribute__((ext_vector_type(8)))  float  v8f;

#define VOCAB 10000
#define EMB   256
#define HID   512
#define G4    2048   // 4*HID
#define BB    32     // batch
#define TT    128    // seq len

static __device__ __forceinline__ v8f wmma_bf16(v16bf a, v16bf b, v8f c) {
    return __builtin_amdgcn_wmma_f32_16x16x32_bf16(false, a, false, b, (short)0, c, false, false);
}

// ---------------- CDNA5 async global->LDS copy (ASYNCcnt-tracked) ----------------
static __device__ __forceinline__ void async_b128(const void* g, void* l) {
#if __has_builtin(__builtin_amdgcn_global_load_async_to_lds_b128)
    typedef int v4i __attribute__((vector_size(4 * sizeof(int))));
    __builtin_amdgcn_global_load_async_to_lds_b128(
        (__attribute__((address_space(1))) v4i*)(void*)g,
        (__attribute__((address_space(3))) v4i*)(void*)l, 0, 0);
#else
    unsigned loff = (unsigned)(size_t)(__attribute__((address_space(3))) void*)l;
    asm volatile("global_load_async_to_lds_b128 %0, %1, off"
                 :: "v"(loff), "v"((unsigned long long)(size_t)g) : "memory");
#endif
}

static __device__ __forceinline__ void wait_async0() {
#if __has_builtin(__builtin_amdgcn_s_wait_asynccnt)
    __builtin_amdgcn_s_wait_asynccnt(0);
#else
    asm volatile("s_wait_asynccnt 0x0" ::: "memory");
#endif
}

// 16-bit A-matrix (16x32) per-lane K index for element e (0..15), lane-half hi.
// lanes 0-15: e0-15 -> K0-15 ; lanes 16-31: K16-31
static __device__ __forceinline__ int ka_map(int hi, int e) {
    return hi * 16 + e;
}

// B-matrix (32x16): lanes 0-15 hold K=0..15 (e = K), lanes 16-31 hold K=16..31.
// swizzled store offset for element (k, n) of a K x N matrix:
static __device__ __forceinline__ size_t swz_off(int k, int n, int K) {
    int kt = k >> 5, kin = k & 31;
    int nt = n >> 4, nin = n & 15;
    int hi = kin >> 4, e = kin & 15;
    int lane = nin + hi * 16;
    int KT = K >> 5;
    return (((size_t)(nt * KT + kt) * 32 + lane) * 16 + e);
}

// Load A fragment (16x32 bf16) from a row-major [M][lda] buffer (global or LDS).
// Per lane this is two contiguous 16B runs -> compiler merges to 2x b128 / ds_load_b128.
static __device__ __forceinline__ v16bf load_a_rowmajor(const __bf16* A, int lda,
                                                        int m0, int k0, int lane) {
    int m  = m0 + (lane & 15);
    int hi = lane >> 4;
    v16bf r;
#pragma unroll
    for (int e = 0; e < 16; ++e)
        r[e] = A[(size_t)m * lda + k0 + ka_map(hi, e)];
    return r;
}

// Load B fragment from pre-swizzled buffer: 16 contiguous bf16 per lane (2 x b128)
static __device__ __forceinline__ v16bf load_b_swz(const __bf16* Bs, int KT,
                                                   int kt, int nt, int lane) {
    const __bf16* p = Bs + (((size_t)(nt * KT + kt) * 32 + lane) * 16);
    return *reinterpret_cast<const v16bf*>(p);
}

static __device__ __forceinline__ float sigm(float x) { return 1.0f / (1.0f + expf(-x)); }

// ---------------- weight convert + swizzle (f32 -> bf16 fragment layout) ----------------
__global__ void convert_swz_kernel(const float* __restrict__ W, __bf16* __restrict__ dst,
                                   int K, int N) {
    size_t idx = (size_t)blockIdx.x * 256 + threadIdx.x;
    size_t total = (size_t)K * N;
    if (idx >= total) return;
    int k = (int)(idx / N), n = (int)(idx % N);
    dst[swz_off(k, n, K)] = (__bf16)W[idx];
}

// ---------------- embedding gather: X[s*32+b][k] = emb[ids[b][s]][k] (bf16) -------------
__global__ void embed_kernel(const int* __restrict__ ids, const float* __restrict__ emb,
                             __bf16* __restrict__ X, int nrows) {
    int m = blockIdx.x;
    if (m >= nrows) return;
    int s = m >> 5, b = m & 31;
    int tok = ids[b * TT + s];
    const float* e = emb + (size_t)tok * EMB;
    for (int k = threadIdx.x; k < EMB; k += blockDim.x)
        X[(size_t)m * EMB + k] = (__bf16)e[k];
}

// ---------------- block-cooperative WMMA GEMM with async A-tile staging ----------------
// C[M][N] = A[M][K](bf16 row-major) @ Bswz + bias.
// Block = 8 waves sharing one M-tile: A tile (16 x K) staged to LDS once via
// global_load_async_to_lds_b128, then each wave computes one 16x16 tile of an 8-wide
// N strip. B fragments are double-buffered in VGPRs (issue kt+1 load before kt WMMA)
// so the backend can overlap one B fetch with each matrix op; L2 prefetch runs kt+2.
__global__ __launch_bounds__(256) void gemm_bias_kernel(
    const __bf16* __restrict__ A, const __bf16* __restrict__ Bswz,
    const float* __restrict__ bias, float* __restrict__ C,
    int M, int N, int K) {
    __shared__ __align__(16) __bf16 Atile[16 * HID];   // 16 KB (max K = 512)
    int tid  = threadIdx.x;
    int lane = tid & 31, wv = tid >> 5;
    int ntN = N >> 4;
    int NT8 = (ntN + 7) >> 3;
    int mt  = blockIdx.x / NT8;
    int nt  = (blockIdx.x % NT8) * 8 + wv;

    // Stage the 16 x K A tile (contiguous 32*K bytes of row-major A) into LDS.
    const char* Ag = (const char*)(A + (size_t)mt * 16 * K);
    char* Al = (char*)Atile;
    int nchunks = 2 * K;                                // (16*K*2B)/16B per b128
    for (int c = tid; c < nchunks; c += 256)
        async_b128(Ag + (size_t)c * 16, Al + (size_t)c * 16);
    wait_async0();
    __syncthreads();

    if (nt < ntN) {                                     // wave-uniform: EXEC all-ones
        int KT = K >> 5;
        v8f acc = {};
        v16bf bcur = load_b_swz(Bswz, KT, 0, nt, lane);
        for (int kt = 0; kt < KT; ++kt) {
            v16bf a = load_a_rowmajor(Atile, K, 0, kt * 32, lane);   // LDS (ds_load)
            v16bf bnxt = bcur;
            if (kt + 1 < KT) {
                bnxt = load_b_swz(Bswz, KT, kt + 1, nt, lane);       // in flight past WMMA
                if (kt + 2 < KT)
                    __builtin_prefetch(Bswz + ((size_t)(nt * KT + kt + 2) * 32 + lane) * 16, 0, 0);
            }
            acc = wmma_bf16(a, bcur, acc);
            bcur = bnxt;
        }
        // f32 C/D layout: VGPR r -> M = r + 8*(lane>=16), N = lane&15
        int hi = lane >> 4, nn = nt * 16 + (lane & 15);
        float bv = bias[nn];
#pragma unroll
        for (int r = 0; r < 8; ++r) {
            int m = mt * 16 + r + hi * 8;
            C[(size_t)m * N + nn] = acc[r] + bv;
        }
    }
}

// ---------------- persistent LSTM: 255 sequential steps on one WGP ---------------------
// 1024 threads = 32 waves. Wave w owns hid-tile nh = w; groups gi in {0,1} are M-tiles.
// c stays in accumulator VGPRs across encoder AND decoder; h staged in LDS as bf16.
__global__ __launch_bounds__(1024, 1) void lstm_kernel(
    const float* __restrict__ Genc, const float* __restrict__ Gdec,
    const __bf16* __restrict__ WhhE, const __bf16* __restrict__ WhhD,
    __bf16* __restrict__ hs /* [4064][HID] bf16 */) {
    __shared__ __bf16 hsh[BB][HID];          // 32 KB
    int tid = threadIdx.x;
    int lane = tid & 31, w = tid >> 5;
    for (int i = tid; i < BB * HID; i += 1024)
        (&hsh[0][0])[i] = (__bf16)0.0f;
    __syncthreads();

    v8f c_acc[2] = {};                       // cell state, lives in VGPRs for all 255 steps
    const int KT = HID / 32;                 // 16

    for (int step = 0; step < 2 * TT - 1; ++step) {
        bool dec = step >= TT;
        int t = dec ? step - TT : step;
        const float*  G = dec ? Gdec : Genc;
        const __bf16* W = dec ? WhhD : WhhE;

        v8f acc[2][4] = {};                  // (M-tile gi) x (gate i,f,g,o)
        for (int kt = 0; kt < KT; ++kt) {
            v16bf a0 = load_a_rowmajor(&hsh[0][0], HID, 0,  kt * 32, lane);
            v16bf a1 = load_a_rowmajor(&hsh[0][0], HID, 16, kt * 32, lane);
#pragma unroll
            for (int q = 0; q < 4; ++q) {    // gate q columns: N = q*512 + w*16 ...
                int nt = q * 32 + w;
                v16bf b = load_b_swz(W, KT, kt, nt, lane);
                acc[0][q] = wmma_bf16(a0, b, acc[0][q]);
                acc[1][q] = wmma_bf16(a1, b, acc[1][q]);
            }
        }
        __syncthreads();                     // all reads of h done before overwrite

        int hi = lane >> 4, nn = lane & 15;
        int col = w * 16 + nn;               // hidden column this lane owns
#pragma unroll
        for (int gi = 0; gi < 2; ++gi) {
#pragma unroll
            for (int r = 0; r < 8; ++r) {
                int m = gi * 16 + r + hi * 8;
                size_t gb = ((size_t)(t * BB + m)) * G4;
                float iv = acc[gi][0][r] + G[gb + 0 * HID + col];
                float fv = acc[gi][1][r] + G[gb + 1 * HID + col];
                float gv = acc[gi][2][r] + G[gb + 2 * HID + col];
                float ov = acc[gi][3][r] + G[gb + 3 * HID + col];
                float c = sigm(fv) * c_acc[gi][r] + sigm(iv) * tanhf(gv);
                float h = sigm(ov) * tanhf(c);
                c_acc[gi][r] = c;
                hsh[m][col] = (__bf16)h;
                if (dec) hs[((size_t)(t * BB + m)) * HID + col] = (__bf16)h;
            }
        }
        __syncthreads();
    }
}

// ---------------- fused scatter/gumbel-softmax/straight-through finalize ---------------
// out row (b,t): mask==0 -> one-hot(token); mask==1,t>=1 -> row = logits[t-1,b]; t==0 -> rand_init[0,b].
// gumbel row index is p = exclusive-cumsum(mask)[b,t]. gs == (hard - y) + y.
__global__ void finalize_kernel(const int* __restrict__ ids, const int* __restrict__ mask,
                                const float* __restrict__ logits,
                                const float* __restrict__ rand_init,
                                const float* __restrict__ gumbel,
                                float* __restrict__ out) {
    int row = blockIdx.x;                    // 0..4095 = b*T + t
    int b = row >> 7, t = row & 127;
    int tid = threadIdx.x;
    float* orow = out + (size_t)row * VOCAB;

    if (mask[b * TT + t] == 0) {
        int tok = ids[b * TT + t];
        for (int j = tid; j < VOCAB; j += 256) orow[j] = (j == tok) ? 1.0f : 0.0f;
        return;
    }
    int p = 0;                               // exclusive prefix of mask along time
    for (int u = 0; u < t; ++u) p += mask[b * TT + u];

    const float* src = (t == 0) ? (rand_init + (size_t)b * VOCAB)
                                : (logits + ((size_t)(t - 1) * BB + b) * VOCAB);
    const float* gum = gumbel + ((size_t)p * BB + b) * VOCAB;

    __shared__ float z[VOCAB];               // 40 KB
    __shared__ float red[256];
    __shared__ int   redi[256];

    float lmax = -INFINITY; int larg = 0x7fffffff;
    for (int j = tid; j < VOCAB; j += 256) {
        float zz = src[j] + gum[j];
        z[j] = zz;
        if (zz > lmax) { lmax = zz; larg = j; }
    }
    red[tid] = lmax; redi[tid] = larg;
    __syncthreads();
    for (int s2 = 128; s2 > 0; s2 >>= 1) {
        if (tid < s2) {
            float v2 = red[tid + s2]; int i2 = redi[tid + s2];
            if (v2 > red[tid] || (v2 == red[tid] && i2 < redi[tid])) { red[tid] = v2; redi[tid] = i2; }
        }
        __syncthreads();
    }
    float m1 = red[0]; int amax = redi[0];
    __syncthreads();
    float lsum = 0.0f;
    for (int j = tid; j < VOCAB; j += 256) {
        float e = expf((z[j] - m1) * 2.0f);  // /TEMP, TEMP = 0.5
        z[j] = e; lsum += e;
    }
    red[tid] = lsum;
    __syncthreads();
    for (int s2 = 128; s2 > 0; s2 >>= 1) { if (tid < s2) red[tid] += red[tid + s2]; __syncthreads(); }
    float inv = 1.0f / red[0];
    for (int j = tid; j < VOCAB; j += 256) {
        float y = z[j] * inv;
        float hard = (j == amax) ? 1.0f : 0.0f;
        orow[j] = (hard - y) + y;            // == reference's stop_gradient trick, bit-faithful
    }
}

// ---------------------------------------------------------------------------------------
extern "C" void kernel_launch(void* const* d_in, const int* in_sizes, int n_in,
                              void* d_out, int out_size, void* d_ws, size_t ws_size,
                              hipStream_t stream) {
    (void)in_sizes; (void)n_in; (void)out_size; (void)ws_size;
    const int*   ids    = (const int*)d_in[0];
    const int*   mask   = (const int*)d_in[1];
    const float* encEmb = (const float*)d_in[2];
    const float* encWih = (const float*)d_in[3];
    const float* encWhh = (const float*)d_in[4];
    const float* encB   = (const float*)d_in[5];
    const float* decEmb = (const float*)d_in[6];
    const float* decWih = (const float*)d_in[7];
    const float* decWhh = (const float*)d_in[8];
    const float* decB   = (const float*)d_in[9];
    const float* fcW    = (const float*)d_in[10];
    const float* fcB    = (const float*)d_in[11];
    const float* randI  = (const float*)d_in[12];
    const float* gumb   = (const float*)d_in[13];
    float* out = (float*)d_out;

    char* ws = (char*)d_ws;
    size_t off = 0;
    auto alloc = [&](size_t bytes) -> char* {
        char* p = ws + off;
        off = (off + bytes + 255) & ~(size_t)255;
        return p;
    };
    const int MD = (TT - 1) * BB;            // 4064 decoder rows (multiple of 16)
    __bf16* wihE = (__bf16*)alloc((size_t)EMB * G4 * 2);
    __bf16* whhE = (__bf16*)alloc((size_t)HID * G4 * 2);
    __bf16* wihD = (__bf16*)alloc((size_t)EMB * G4 * 2);
    __bf16* whhD = (__bf16*)alloc((size_t)HID * G4 * 2);
    __bf16* fcWs = (__bf16*)alloc((size_t)HID * VOCAB * 2);
    __bf16* Xenc = (__bf16*)alloc((size_t)TT * BB * EMB * 2);
    __bf16* Xdec = (__bf16*)alloc((size_t)MD * EMB * 2);
    float*  Genc = (float*)alloc((size_t)TT * BB * G4 * 4);
    float*  Gdec = (float*)alloc((size_t)MD * G4 * 4);
    __bf16* hsB  = (__bf16*)alloc((size_t)MD * HID * 2);
    float*  logi = (float*)alloc((size_t)MD * VOCAB * 4);

    // 1. weights -> bf16 fragment-swizzled
    convert_swz_kernel<<<(EMB * G4 + 255) / 256, 256, 0, stream>>>(encWih, wihE, EMB, G4);
    convert_swz_kernel<<<(HID * G4 + 255) / 256, 256, 0, stream>>>(encWhh, whhE, HID, G4);
    convert_swz_kernel<<<(EMB * G4 + 255) / 256, 256, 0, stream>>>(decWih, wihD, EMB, G4);
    convert_swz_kernel<<<(HID * G4 + 255) / 256, 256, 0, stream>>>(decWhh, whhD, HID, G4);
    convert_swz_kernel<<<(HID * VOCAB + 255) / 256, 256, 0, stream>>>(fcW, fcWs, HID, VOCAB);

    // 2. embeddings
    embed_kernel<<<TT * BB, 256, 0, stream>>>(ids, encEmb, Xenc, TT * BB);
    embed_kernel<<<MD, 256, 0, stream>>>(ids, decEmb, Xdec, MD);

    // 3. hoisted input-gate GEMMs over all timesteps (WMMA bf16, async A staging)
    {
        int ntN = G4 / 16, NT8 = (ntN + 7) / 8;
        gemm_bias_kernel<<<(TT * BB / 16) * NT8, 256, 0, stream>>>(
            Xenc, wihE, encB, Genc, TT * BB, G4, EMB);
        gemm_bias_kernel<<<(MD / 16) * NT8, 256, 0, stream>>>(
            Xdec, wihD, decB, Gdec, MD, G4, EMB);
    }

    // 4. persistent recurrence (encoder 128 steps + decoder 127 steps, one WGP)
    lstm_kernel<<<1, 1024, 0, stream>>>(Genc, Gdec, whhE, whhD, hsB);

    // 5. fc GEMM (FLOP-dominant, WMMA bf16, f32 accumulate, async A staging)
    {
        int ntN = VOCAB / 16, NT8 = (ntN + 7) / 8;
        gemm_bias_kernel<<<(MD / 16) * NT8, 256, 0, stream>>>(
            hsB, fcWs, fcB, logi, MD, VOCAB, HID);
    }

    // 6. fused gather/gumbel-softmax/straight-through -> output
    finalize_kernel<<<BB * TT, 256, 0, stream>>>(ids, mask, logi, randI, gumb, out);
}
